// VQGate_61701500175229
// MI455X (gfx1250) — matrix operand order repaired and checked by reading, not compiled, not run on hardware
//
#include <hip/hip_runtime.h>

// VQ gate forward:
//   out[t,:] = target[t,:] * (1 + E[argmax_k (z[t,:] . normalize(codebook[k,:])), :])
// (softmax / straight-through collapse to a hard argmax in forward; z-normalization
//  and TAU are positive monotone scalings and cannot change the argmax.)

typedef _Float16 h4   __attribute__((ext_vector_type(4)));
typedef _Float16 v8h  __attribute__((ext_vector_type(8)));
typedef _Float16 v16h __attribute__((ext_vector_type(16)));
typedef float    v8f  __attribute__((ext_vector_type(8)));

#define DIM           1024      // d_in == d_g
#define KCODES        512
#define NTOK          16384     // B*N = 4*4096
#define RS            1032      // padded LDS row stride in halves (2064 B)
#define WAVES         8
#define ROWS_PER_WAVE 16
#define TOK_PER_BLOCK 128

// ---------------------------------------------------------------------------
// Kernel 1: Cf = l2-normalize(codebook) rows, stored as f16 in workspace.
// ---------------------------------------------------------------------------
__global__ __launch_bounds__(256) void vq_norm_codebook(const float* __restrict__ cb,
                                                        _Float16* __restrict__ cf) {
    const int row  = blockIdx.x;
    const int t    = threadIdx.x;          // 256 threads, 4 floats each
    const int lane = t & 31;
    const int wid  = t >> 5;

    const float4* s4 = (const float4*)(cb + (size_t)row * DIM);
    float4 v = s4[t];
    float ss = v.x * v.x + v.y * v.y + v.z * v.z + v.w * v.w;
#pragma unroll
    for (int m = 1; m < 32; m <<= 1) ss += __shfl_xor(ss, m, 32);

    __shared__ float red[WAVES];
    if (lane == 0) red[wid] = ss;
    __syncthreads();
    float tot = 0.0f;
#pragma unroll
    for (int i = 0; i < WAVES; ++i) tot += red[i];

    const float s = 1.0f / fmaxf(sqrtf(tot), 1e-12f);   // matches F.normalize eps
    h4 o;
    o[0] = (_Float16)(v.x * s);
    o[1] = (_Float16)(v.y * s);
    o[2] = (_Float16)(v.z * s);
    o[3] = (_Float16)(v.w * s);
    *(h4*)(cf + (size_t)row * DIM + (size_t)t * 4) = o;
}

// ---------------------------------------------------------------------------
// Fragment loaders (ISA 7.12.2 layouts).
// A 16x32 f16: lane holds row M=lane%16; two contiguous b128 chunks per half.
// B 32x16 f16: lane holds column N=lane%16 -> codebook row; same K chunks as A.
// ---------------------------------------------------------------------------
__device__ __forceinline__ v16h load_a_at(const _Float16* __restrict__ ap, int half) {
    v8h lo = *(const v8h*)(ap + half * 8);
    v8h hi = *(const v8h*)(ap + 16 + half * 8);
    v16h a;
#pragma unroll
    for (int i = 0; i < 8; ++i) { a[i] = lo[i]; a[i + 8] = hi[i]; }
    return a;
}

__device__ __forceinline__ v16h load_b_at(const _Float16* __restrict__ bp, int t, int half) {
    const _Float16* p = bp + (size_t)t * 16 * DIM + half * 8;
    v8h lo = *(const v8h*)p;
    v8h hi = *(const v8h*)(p + 16);
    v16h b;
#pragma unroll
    for (int i = 0; i < 8; ++i) { b[i] = lo[i]; b[i + 8] = hi[i]; }
    return b;
}

// ---------------------------------------------------------------------------
// Kernel 2: per-token argmax over codes via v_wmma_f32_16x16x32_f16, then
//           out = target * (1 + E[idx]).
// One wave owns a 16-token tile; 8 waves per block -> 128 tokens per block.
// dk trip count is a runtime argument so the compiler cannot fully unroll the
// loop: the explicit 1-deep rotating buffer (a/b -> an/bn) survives, so every
// WMMA consumes operands loaded a full iteration earlier.
// ---------------------------------------------------------------------------
__global__ __launch_bounds__(256) void vq_gate_main(const float* __restrict__ z,
                                                    const float* __restrict__ target,
                                                    const float* __restrict__ E,
                                                    const _Float16* __restrict__ cf,
                                                    float* __restrict__ out,
                                                    int nchunks) {
    extern __shared__ _Float16 zs[];

    const int tid  = threadIdx.x;
    const int lane = tid & 31;
    const int w    = tid >> 5;
    const int half = lane >> 4;   // 0: lanes 0-15, 1: lanes 16-31
    const int n    = lane & 15;   // WMMA B/C column index (and A row M)
    const int tokenbase = blockIdx.x * TOK_PER_BLOCK + w * ROWS_PER_WAVE;

    _Float16* zw = zs + (size_t)w * ROWS_PER_WAVE * RS;

    // ---- Stage this wave's 16x1024 z tile to LDS as f16 (read once from HBM)
    const float4* zg = (const float4*)(z + (size_t)tokenbase * DIM);
    for (int i = lane; i < ROWS_PER_WAVE * (DIM / 4); i += 32) {
        const int r  = i >> 8;        // DIM/4 == 256 float4 per row
        const int c4 = i & 255;
        float4 v = zg[r * (DIM / 4) + c4];
        h4 o;
        o[0] = (_Float16)v.x; o[1] = (_Float16)v.y;
        o[2] = (_Float16)v.z; o[3] = (_Float16)v.w;
        *(h4*)(zw + r * RS + c4 * 4) = o;
    }
    __syncthreads();   // uniform across all 8 waves

    // ---- Running per-lane argmax. C-layout: lane holds N=n, VGPR v holds
    //      M = v + 8*half; candidate code index = kc*16 + n (same for all v).
    float bestv[8];
    int   besti[8];
#pragma unroll
    for (int v = 0; v < 8; ++v) { bestv[v] = -3.0e38f; besti[v] = 0; }

    const _Float16* arow = zw + n * RS;   // A: lane reads token row M = n

    for (int g = 0; g < 8; ++g) {         // 8 groups of 4 code tiles = 512 codes
        const _Float16* bbase = cf + (size_t)(g * 64 + n) * DIM;
        v8f acc[4] = {};

        // Pipeline prologue: dk = 0 operands
        v16h a = load_a_at(arow, half);
        v16h b[4];
#pragma unroll
        for (int t = 0; t < 4; ++t) b[t] = load_b_at(bbase, t, half);

        for (int dk = 0; dk < nchunks - 1; ++dk) {
            // Issue next iteration's loads first (independent registers).
            const _Float16* apn = arow + (dk + 1) * 32;
            const _Float16* bpn = bbase + (dk + 1) * 32;
            v16h an = load_a_at(apn, half);
            v16h bn[4];
#pragma unroll
            for (int t = 0; t < 4; ++t) bn[t] = load_b_at(bpn, t, half);

            // Warm WGP$ ~4 dk ahead (each lane one 64B row-chunk, alternating
            // row parity per iteration). locality=3 -> nearest-scope prefetch.
            __builtin_prefetch((const char*)(cf
                                + (size_t)(g * 64 + (((lane << 1) | (dk & 1)) & 63)) * DIM)
                                + (size_t)((dk + 4) & 31) * 64, 0, 3);

#pragma unroll
            for (int t = 0; t < 4; ++t)
                acc[t] = __builtin_amdgcn_wmma_f32_16x16x32_f16(
                    false, a, false, b[t], (short)0, acc[t], false, false);

            a = an;
#pragma unroll
            for (int t = 0; t < 4; ++t) b[t] = bn[t];
        }
        // Pipeline epilogue: last dk
#pragma unroll
        for (int t = 0; t < 4; ++t)
            acc[t] = __builtin_amdgcn_wmma_f32_16x16x32_f16(
                false, a, false, b[t], (short)0, acc[t], false, false);

#pragma unroll
        for (int t = 0; t < 4; ++t) {
            const int idxc = g * 64 + t * 16 + n;
#pragma unroll
            for (int v = 0; v < 8; ++v) {
                const float val = acc[t][v];
                if (val > bestv[v]) { bestv[v] = val; besti[v] = idxc; }
            }
        }
    }

    // ---- Cross-lane argmax over the 16 columns within each half-wave.
    //      Ties break to the lower code index (matches jnp.argmax).
#pragma unroll
    for (int m = 1; m < 16; m <<= 1) {
#pragma unroll
        for (int v = 0; v < 8; ++v) {
            const float ov = __shfl_xor(bestv[v], m, 32);
            const int   oi = __shfl_xor(besti[v], m, 32);
            if (ov > bestv[v] || (ov == bestv[v] && oi < besti[v])) {
                bestv[v] = ov; besti[v] = oi;
            }
        }
    }

    // ---- Route: lane handling token row r = n needs besti[r%8] from half r/8.
    int myidx = 0;
#pragma unroll
    for (int v = 0; v < 8; ++v) {
        const int i0 = __shfl(besti[v], 0, 32);    // rows 0..7  (half 0)
        const int i1 = __shfl(besti[v], 16, 32);   // rows 8..15 (half 1)
        if (n == v)     myidx = i0;
        if (n == v + 8) myidx = i1;
    }

    // ---- Epilogue: two lanes per token row, 512 floats each (float4 vectors).
    const int token = tokenbase + n;
    const float4* tp = (const float4*)(target + (size_t)token * DIM) + half * 128;
    const float4* ep = (const float4*)(E + (size_t)myidx * DIM) + half * 128;
    float4*       op = (float4*)(out + (size_t)token * DIM) + half * 128;
#pragma unroll 4
    for (int i = 0; i < 128; ++i) {
        const float4 t4 = tp[i];
        const float4 e4 = ep[i];
        float4 o;
        o.x = t4.x * (1.0f + e4.x);
        o.y = t4.y * (1.0f + e4.y);
        o.z = t4.z * (1.0f + e4.z);
        o.w = t4.w * (1.0f + e4.w);
        op[i] = o;
    }
}

// ---------------------------------------------------------------------------
extern "C" void kernel_launch(void* const* d_in, const int* in_sizes, int n_in,
                              void* d_out, int out_size, void* d_ws, size_t ws_size,
                              hipStream_t stream) {
    const float* z      = (const float*)d_in[0];   // [4,4096,1024] f32
    const float* target = (const float*)d_in[1];   // [4,4096,1024] f32
    const float* cb     = (const float*)d_in[2];   // [512,1024]    f32
    const float* E      = (const float*)d_in[3];   // [512,1024]    f32
    float*       out    = (float*)d_out;           // [4,4096,1024] f32
    _Float16*    cf     = (_Float16*)d_ws;         // 512*1024 f16 = 1 MB scratch

    vq_norm_codebook<<<KCODES, 256, 0, stream>>>(cb, cf);

    const size_t lds_bytes = (size_t)WAVES * ROWS_PER_WAVE * RS * sizeof(_Float16);
    vq_gate_main<<<NTOK / TOK_PER_BLOCK, 256, lds_bytes, stream>>>(z, target, E, cf, out,
                                                                   DIM / 32);
}
